// TFMultiHeadAttention_35613868819148
// MI455X (gfx1250) — compile-verified, matrix operand order checked by reading, not run on hardware
//
#include <hip/hip_runtime.h>
#include <hip/hip_bf16.h>
#include <stdint.h>

// ---------------------------------------------------------------------------
// MHA forward for MI455X (gfx1250): bf16 WMMA GEMMs + flash-style attention
// with async global->LDS double-buffered K/V staging.
// BS=2, QLEN=2048, DIM=1024, NH=16, HD=64.
// ---------------------------------------------------------------------------

#define BS    2
#define QLEN  2048
#define DIM   1024
#define NH    16
#define HD    64
#define MTOT  (BS * QLEN)          // 4096 rows of X

typedef __attribute__((ext_vector_type(16))) __bf16 bf16x16;
typedef __attribute__((ext_vector_type(8)))  float  f32x8;
typedef __attribute__((ext_vector_type(4)))  int    i32x4;

union FragU {
    bf16x16  v;
    uint32_t u[8];
};

static __device__ __forceinline__ uint16_t f2bf(float f) {
    uint32_t u = __float_as_uint(f);
    u = (u + 0x7FFFu + ((u >> 16) & 1u)) >> 16;   // round-to-nearest-even
    return (uint16_t)u;
}

// K offset within a 32-wide chunk for fragment dword j, per the CDNA5
// 16-bit A/B matrix VGPR layout (lanes 0-15 vs 16-31 = hi half).
static __device__ __forceinline__ int frag_k(int j, int hi) {
    return (j < 4) ? (j * 2 + hi * 8) : (16 + (j - 4) * 2 + hi * 8);
}

static __device__ __forceinline__ f32x8 wmma_bf16(const FragU& a, const FragU& b, f32x8 c) {
    return __builtin_amdgcn_wmma_f32_16x16x32_bf16(false, a.v, false, b.v,
                                                   (short)0, c, false, false);
}

// ---------------------------------------------------------------------------
// Async global -> LDS staging (GLOBAL_LOAD_ASYNC_TO_LDS_B128, ASYNCcnt),
// with a synchronous VGPR-staged fallback if the builtin is unavailable.
// Builtin signature (from probe): void(AS1 int4*, AS3 int4*, imm off, imm cpol)
// ---------------------------------------------------------------------------
#define AS1 __attribute__((address_space(1)))
#define AS3 __attribute__((address_space(3)))

#if __has_builtin(__builtin_amdgcn_global_load_async_to_lds_b128)
#define HAVE_ASYNC 1
#else
#define HAVE_ASYNC 0
#endif

static __device__ __forceinline__ void ld_async_b128(const uint16_t* g, uint16_t* l) {
#if HAVE_ASYNC
    __builtin_amdgcn_global_load_async_to_lds_b128((AS1 i32x4*)g, (AS3 i32x4*)l, 0, 0);
#else
    *(uint4*)l = *(const uint4*)g;
#endif
}

template <int N>
static __device__ __forceinline__ void wait_async() {
#if HAVE_ASYNC
#if __has_builtin(__builtin_amdgcn_s_wait_asynccnt)
    __builtin_amdgcn_s_wait_asynccnt(N);
#else
    asm volatile("s_wait_asynccnt %0" :: "i"(N) : "memory");
#endif
#endif
}

// ---------------------------------------------------------------------------
// Row-of-16 butterfly reductions. Prefer v_permlane16_b32 (pure VALU, no DS
// pipe) over __shfl_xor (which lowers to ds_bpermute_b32 on gfx1250).
// Lane-select constants implement lane -> lane^o within each 16-lane row.
// ---------------------------------------------------------------------------
#if __has_builtin(__builtin_amdgcn_permlane16)
#define PL16(u, lo, hi) __builtin_amdgcn_permlane16((u), (u), (lo), (hi), false, false)
#define HAVE_PL16 1
#elif __has_builtin(__builtin_amdgcn_permlane16_b32)
#define PL16(u, lo, hi) __builtin_amdgcn_permlane16_b32((u), (u), (lo), (hi), false, false)
#define HAVE_PL16 1
#else
#define HAVE_PL16 0
#endif

#if HAVE_PL16
static __device__ __forceinline__ float pl16_xor(float v, uint32_t lo, uint32_t hi) {
    return __uint_as_float(PL16(__float_as_uint(v), lo, hi));
}
static __device__ __forceinline__ float row16_max(float v) {
    v = fmaxf(v, pl16_xor(v, 0x67452301u, 0xEFCDAB89u));   // xor 1
    v = fmaxf(v, pl16_xor(v, 0x54761032u, 0xDCFE98BAu));   // xor 2
    v = fmaxf(v, pl16_xor(v, 0x32107654u, 0xBA98FEDCu));   // xor 4
    v = fmaxf(v, pl16_xor(v, 0xFEDCBA98u, 0x76543210u));   // xor 8
    return v;
}
static __device__ __forceinline__ float row16_sum(float v) {
    v += pl16_xor(v, 0x67452301u, 0xEFCDAB89u);
    v += pl16_xor(v, 0x54761032u, 0xDCFE98BAu);
    v += pl16_xor(v, 0x32107654u, 0xBA98FEDCu);
    v += pl16_xor(v, 0xFEDCBA98u, 0x76543210u);
    return v;
}
#else
static __device__ __forceinline__ float row16_max(float v) {
#pragma unroll
    for (int o = 1; o < 16; o <<= 1) v = fmaxf(v, __shfl_xor(v, o, 32));
    return v;
}
static __device__ __forceinline__ float row16_sum(float v) {
#pragma unroll
    for (int o = 1; o < 16; o <<= 1) v += __shfl_xor(v, o, 32);
    return v;
}
#endif

// ---------------------------------------------------------------------------
// Kernel: fp32 -> bf16 elementwise convert
// ---------------------------------------------------------------------------
__global__ __launch_bounds__(256) void cvt_bf16_kernel(const float* __restrict__ in,
                                                       uint16_t* __restrict__ out,
                                                       int n) {
    int i = blockIdx.x * 256 + threadIdx.x;
    if (i < n) out[i] = f2bf(in[i]);
}

// ---------------------------------------------------------------------------
// Kernel: W[k][n] (fp32, 1024x1024) -> Wt[n][k] (bf16), LDS-tiled transpose
// ---------------------------------------------------------------------------
__global__ __launch_bounds__(256) void transpose_cvt_kernel(const float* __restrict__ W,
                                                            uint16_t* __restrict__ Wt) {
    __shared__ float tile[16][17];
    const int c = threadIdx.x & 15;
    const int r = threadIdx.x >> 4;
    const int n0 = blockIdx.x * 16;
    const int k0 = blockIdx.y * 16;
    tile[r][c] = W[(size_t)(k0 + r) * DIM + (n0 + c)];
    __syncthreads();
    Wt[(size_t)(n0 + r) * DIM + (k0 + c)] = f2bf(tile[c][r]);
}

// ---------------------------------------------------------------------------
// Templated WMMA GEMM: Y = A[4096x1024](bf16) * Bt'[1024x1024](bf16, stored
// row-major over output column N) + bias.
//   MODE 0: Q  -> (acc+bias)*0.125, layout [b,h,q,hd] bf16
//   MODE 1: K  -> (acc+bias),       layout [b,h,q,hd] bf16
//   MODE 2: Vt -> (acc+bias),       layout [b,h,hd,q] bf16 (transposed)
//   MODE 3: out-> (acc+bias),       layout [4096,1024] fp32
// Block = 256 threads = 8 waves arranged 2(M) x 4(N); wave = 32x32 output
// tile = 2x2 WMMA tiles (4 fragment loads feed 4 WMMAs).
// ---------------------------------------------------------------------------
template <int MODE>
__global__ __launch_bounds__(256) void gemm_bf16_kernel(const uint16_t* __restrict__ A,
                                                        const uint16_t* __restrict__ Bt,
                                                        const float* __restrict__ bias,
                                                        void* __restrict__ Out) {
    const int lane = threadIdx.x & 31;
    const int wave = threadIdx.x >> 5;
    const int hi   = lane >> 4;
    const int l15  = lane & 15;
    const int wm   = wave & 1;
    const int wn   = wave >> 1;
    const int Mbase = blockIdx.y * 64 + wm * 32;
    const int Nbase = blockIdx.x * 128 + wn * 32;

    const uint16_t* ar0 = A  + (size_t)(Mbase + l15) * DIM;
    const uint16_t* ar1 = A  + (size_t)(Mbase + 16 + l15) * DIM;
    const uint16_t* br0 = Bt + (size_t)(Nbase + l15) * DIM;
    const uint16_t* br1 = Bt + (size_t)(Nbase + 16 + l15) * DIM;

    f32x8 acc[2][2];
#pragma unroll
    for (int i = 0; i < 2; ++i)
#pragma unroll
        for (int j = 0; j < 2; ++j)
            acc[i][j] = (f32x8){0.f,0.f,0.f,0.f,0.f,0.f,0.f,0.f};

    for (int k0 = 0; k0 < DIM; k0 += 32) {
        __builtin_prefetch(ar0 + k0 + 64, 0, 1);
        __builtin_prefetch(br0 + k0 + 64, 0, 1);
        FragU a0, a1, b0, b1;
#pragma unroll
        for (int j = 0; j < 8; ++j) {
            const int k = k0 + frag_k(j, hi);
            a0.u[j] = *(const uint32_t*)(ar0 + k);
            a1.u[j] = *(const uint32_t*)(ar1 + k);
            b0.u[j] = *(const uint32_t*)(br0 + k);
            b1.u[j] = *(const uint32_t*)(br1 + k);
        }
        acc[0][0] = wmma_bf16(a0, b0, acc[0][0]);
        acc[0][1] = wmma_bf16(a0, b1, acc[0][1]);
        acc[1][0] = wmma_bf16(a1, b0, acc[1][0]);
        acc[1][1] = wmma_bf16(a1, b1, acc[1][1]);
    }

#pragma unroll
    for (int i = 0; i < 2; ++i) {
#pragma unroll
        for (int j = 0; j < 2; ++j) {
#pragma unroll
            for (int r = 0; r < 8; ++r) {
                const int gm = Mbase + i * 16 + r + hi * 8;  // token index
                const int gn = Nbase + j * 16 + l15;         // output column
                float v = acc[i][j][r] + bias[gn];
                if (MODE == 0 || MODE == 1) {
                    if (MODE == 0) v *= 0.125f;              // 1/sqrt(64)
                    const int bb = gm >> 11, q = gm & (QLEN - 1);
                    const int h = gn >> 6, d = gn & (HD - 1);
                    ((uint16_t*)Out)[(((size_t)(bb * NH + h) * QLEN + q) << 6) + d] = f2bf(v);
                } else if (MODE == 2) {
                    const int bb = gm >> 11, q = gm & (QLEN - 1);
                    const int h = gn >> 6, d = gn & (HD - 1);
                    ((uint16_t*)Out)[(((size_t)(bb * NH + h) * HD + d) << 11) + q] = f2bf(v);
                } else {
                    ((float*)Out)[(size_t)gm * DIM + gn] = v;
                }
            }
        }
    }
}

// ---------------------------------------------------------------------------
// Cooperative async staging of one 64-wide KV tile into LDS.
//   gK: K tile base (rows contiguous, 64 rows x 64 elems)
//   gV: Vt head base (rows = hd, row stride QLEN); column window [kv, kv+64)
// 4 x b128 async loads per thread (2 K + 2 V).
// ---------------------------------------------------------------------------
static __device__ __forceinline__ void stage_tile(const uint16_t* __restrict__ gK,
                                                  const uint16_t* __restrict__ gV,
                                                  uint16_t* lK, uint16_t* lV,
                                                  int tid, int kv) {
#pragma unroll
    for (int c = 0; c < 2; ++c) {
        const int idx = tid * 2 + c;          // 0..511
        const int row = idx >> 3;             // 0..63
        const int seg = (idx & 7) * 8;        // element offset, 16B chunks
        ld_async_b128(gK + (size_t)row * HD + seg,        lK + row * 64 + seg);
        ld_async_b128(gV + (size_t)row * QLEN + kv + seg, lV + row * 64 + seg);
    }
}

// ---------------------------------------------------------------------------
// Flash-style attention: block = 8 waves = 128 q rows of one (b,h). K/V tiles
// (64 kv wide) are async-staged into double-buffered LDS and shared by all
// waves; each wave runs online softmax over its 16 q rows (16 WMMAs / tile).
//   Qb/Kb: [b,h,q,hd] bf16 (Q pre-scaled by 1/sqrt(hd)); Vt: [b,h,hd,q] bf16.
//   ctx:   [b*q, h*hd] bf16 (row-major [4096,1024]).
// ---------------------------------------------------------------------------
__global__ __launch_bounds__(256) void attn_kernel(const uint16_t* __restrict__ Qb,
                                                   const uint16_t* __restrict__ Kb,
                                                   const uint16_t* __restrict__ Vt,
                                                   const float* __restrict__ mask,
                                                   uint16_t* __restrict__ ctx) {
    __shared__ uint16_t Kl[2][64][64];     // 16 KB: K tile, rows = kv
    __shared__ uint16_t Vl[2][64][64];     // 16 KB: V tile, rows = hd
    __shared__ uint16_t Plds[8][16][64];   // 16 KB: per-wave P tiles

    const int tid  = threadIdx.x;
    const int lane = tid & 31;
    const int wave = tid >> 5;
    const int hi   = lane >> 4;
    const int l15  = lane & 15;
    const int b    = blockIdx.z;
    const int h    = blockIdx.y;
    const int qbase = blockIdx.x * 128 + wave * 16;

    const size_t bh = (size_t)(b * NH + h);
    const uint16_t* Qp = Qb + (bh * QLEN + qbase) * HD;
    const uint16_t* Kp = Kb + bh * QLEN * HD;
    const uint16_t* Vp = Vt + bh * HD * QLEN;
    const float*    mp = mask + (size_t)b * QLEN;

    // Q A-fragments: row = l15, two 32-wide K chunks covering hd 0..63
    FragU qa0, qa1;
    {
        const uint16_t* qrow = Qp + (size_t)l15 * HD;
#pragma unroll
        for (int j = 0; j < 8; ++j) {
            const int k = frag_k(j, hi);
            qa0.u[j] = *(const uint32_t*)(qrow + k);
            qa1.u[j] = *(const uint32_t*)(qrow + 32 + k);
        }
    }

    f32x8 oacc[4];
#pragma unroll
    for (int nc = 0; nc < 4; ++nc) oacc[nc] = (f32x8){0.f,0.f,0.f,0.f,0.f,0.f,0.f,0.f};
    float m8[8], l8[8];
#pragma unroll
    for (int r = 0; r < 8; ++r) { m8[r] = -3.0e38f; l8[r] = 0.f; }

    // Preload tile 0 into buffer 0.
    stage_tile(Kp, Vp, &Kl[0][0][0], &Vl[0][0][0], tid, 0);

    for (int it = 0; it < QLEN / 64; ++it) {
        const int kv  = it * 64;
        const int buf = it & 1;

        // Kick off the next tile into the other buffer, then wait for ours:
        // ASYNCcnt decrements in issue order, so <=4 means tile `it` is done.
        if (it + 1 < QLEN / 64) {
            stage_tile(Kp + (size_t)(kv + 64) * HD, Vp,
                       &Kl[buf ^ 1][0][0], &Vl[buf ^ 1][0][0], tid, kv + 64);
            wait_async<4>();
        } else {
            wait_async<0>();
        }
        __syncthreads();

        // ---- S = Q * K^T for 4 n-tiles of 16 kv columns (8 WMMAs) ---------
        f32x8 s[4];
#pragma unroll
        for (int j = 0; j < 4; ++j) {
            FragU klo, khi;
            const uint16_t* kr = &Kl[buf][j * 16 + l15][0];
#pragma unroll
            for (int t = 0; t < 8; ++t) {
                const int k = frag_k(t, hi);
                klo.u[t] = *(const uint32_t*)(kr + k);
                khi.u[t] = *(const uint32_t*)(kr + 32 + k);
            }
            f32x8 z = (f32x8){0.f,0.f,0.f,0.f,0.f,0.f,0.f,0.f};
            z = wmma_bf16(qa0, klo, z);
            s[j] = wmma_bf16(qa1, khi, z);
        }

        float pen[4];
#pragma unroll
        for (int j = 0; j < 4; ++j)
            pen[j] = 1.0e30f * (1.0f - mp[kv + j * 16 + l15]);

        // ---- online softmax; write P straight into per-wave LDS tile ------
#pragma unroll
        for (int r = 0; r < 8; ++r) {
            float v0 = s[0][r] - pen[0];
            float v1 = s[1][r] - pen[1];
            float v2 = s[2][r] - pen[2];
            float v3 = s[3][r] - pen[3];
            float tmax = fmaxf(fmaxf(v0, v1), fmaxf(v2, v3));
            tmax = row16_max(tmax);
            const float mnew = fmaxf(m8[r], tmax);
            const float corr = __expf(m8[r] - mnew);
            const float p0 = __expf(v0 - mnew);
            const float p1 = __expf(v1 - mnew);
            const float p2 = __expf(v2 - mnew);
            const float p3 = __expf(v3 - mnew);
            const float rsum = row16_sum((p0 + p1) + (p2 + p3));
            l8[r] = l8[r] * corr + rsum;
            m8[r] = mnew;
#pragma unroll
            for (int nc = 0; nc < 4; ++nc) oacc[nc][r] *= corr;
            const int row = r + hi * 8;
            Plds[wave][row][l15]      = f2bf(p0);
            Plds[wave][row][16 + l15] = f2bf(p1);
            Plds[wave][row][32 + l15] = f2bf(p2);
            Plds[wave][row][48 + l15] = f2bf(p3);
        }
        asm volatile("s_wait_dscnt 0" ::: "memory");

        // ---- P back as A-fragments (16x64 -> two 16x32 frags) -------------
        FragU pa0, pa1;
#pragma unroll
        for (int t = 0; t < 8; ++t) {
            const int k = frag_k(t, hi);
            pa0.u[t] = *(const uint32_t*)(&Plds[wave][l15][k]);
            pa1.u[t] = *(const uint32_t*)(&Plds[wave][l15][32 + k]);
        }

        // ---- O += P * V (8 WMMAs; V tile in LDS, rows = hd) ---------------
#pragma unroll
        for (int nc = 0; nc < 4; ++nc) {
            FragU vb0, vb1;
            const uint16_t* vrow = &Vl[buf][nc * 16 + l15][0];
#pragma unroll
            for (int t = 0; t < 8; ++t) {
                const int k = frag_k(t, hi);
                vb0.u[t] = *(const uint32_t*)(vrow + k);
                vb1.u[t] = *(const uint32_t*)(vrow + 32 + k);
            }
            oacc[nc] = wmma_bf16(pa0, vb0, oacc[nc]);
            oacc[nc] = wmma_bf16(pa1, vb1, oacc[nc]);
        }
        __syncthreads();   // all waves done with buf before it is re-staged
    }

    // ---- normalize and emit context [token, h*64+d] as bf16 ---------------
#pragma unroll
    for (int r = 0; r < 8; ++r) {
        const float inv = (l8[r] > 0.f) ? (1.0f / l8[r]) : 0.f;
        const int row = r + hi * 8;
        const size_t base = ((size_t)(b * QLEN + qbase + row)) * DIM + h * HD;
#pragma unroll
        for (int nc = 0; nc < 4; ++nc)
            ctx[base + nc * 16 + l15] = f2bf(oacc[nc][r] * inv);
    }
}

// ---------------------------------------------------------------------------
// Host-side launch
// ---------------------------------------------------------------------------
extern "C" void kernel_launch(void* const* d_in, const int* in_sizes, int n_in,
                              void* d_out, int out_size, void* d_ws, size_t ws_size,
                              hipStream_t stream) {
    const float* x    = (const float*)d_in[0];
    const float* mask = (const float*)d_in[1];
    const float* Wq   = (const float*)d_in[2];
    const float* bq   = (const float*)d_in[3];
    const float* Wk   = (const float*)d_in[4];
    const float* bk   = (const float*)d_in[5];
    const float* Wv   = (const float*)d_in[6];
    const float* bv   = (const float*)d_in[7];
    const float* Wo   = (const float*)d_in[8];
    const float* bo   = (const float*)d_in[9];
    float* out = (float*)d_out;

    // Workspace carve-up (bf16 buffers), ~50.3 MB total.
    uint8_t* ws = (uint8_t*)d_ws;
    uint16_t* Xbf = (uint16_t*)ws;  ws += (size_t)MTOT * DIM * 2;
    uint16_t* WqT = (uint16_t*)ws;  ws += (size_t)DIM * DIM * 2;
    uint16_t* WkT = (uint16_t*)ws;  ws += (size_t)DIM * DIM * 2;
    uint16_t* WvT = (uint16_t*)ws;  ws += (size_t)DIM * DIM * 2;
    uint16_t* WoT = (uint16_t*)ws;  ws += (size_t)DIM * DIM * 2;
    uint16_t* Qb  = (uint16_t*)ws;  ws += (size_t)BS * NH * QLEN * HD * 2;
    uint16_t* Kb  = (uint16_t*)ws;  ws += (size_t)BS * NH * QLEN * HD * 2;
    uint16_t* Vt  = (uint16_t*)ws;  ws += (size_t)BS * NH * QLEN * HD * 2;
    uint16_t* ctx = (uint16_t*)ws;  ws += (size_t)MTOT * DIM * 2;

    // 1) convert x -> bf16; transpose-convert weights
    {
        const int n = MTOT * DIM;
        cvt_bf16_kernel<<<(n + 255) / 256, 256, 0, stream>>>(x, Xbf, n);
        dim3 tg(DIM / 16, DIM / 16);
        transpose_cvt_kernel<<<tg, 256, 0, stream>>>(Wq, WqT);
        transpose_cvt_kernel<<<tg, 256, 0, stream>>>(Wk, WkT);
        transpose_cvt_kernel<<<tg, 256, 0, stream>>>(Wv, WvT);
        transpose_cvt_kernel<<<tg, 256, 0, stream>>>(Wo, WoT);
    }

    // 2) QKV projections (WMMA bf16), head-split / transposed layouts
    {
        dim3 grid(DIM / 128, MTOT / 64);
        gemm_bf16_kernel<0><<<grid, 256, 0, stream>>>(Xbf, WqT, bq, Qb);
        gemm_bf16_kernel<1><<<grid, 256, 0, stream>>>(Xbf, WkT, bk, Kb);
        gemm_bf16_kernel<2><<<grid, 256, 0, stream>>>(Xbf, WvT, bv, Vt);
    }

    // 3) fused flash attention (async-staged K/V) -> bf16 context
    {
        dim3 grid(QLEN / 128, NH, BS);
        attn_kernel<<<grid, 256, 0, stream>>>(Qb, Kb, Vt, mask, ctx);
    }

    // 4) output projection -> fp32
    {
        dim3 grid(DIM / 128, MTOT / 64);
        gemm_bf16_kernel<3><<<grid, 256, 0, stream>>>(ctx, WoT, bo, out);
    }
}
